// MemoryEfficientAttention_60748017434898
// MI455X (gfx1250) — compile-verified, matrix operand order checked by reading, not run on hardware
//
#include <hip/hip_runtime.h>
#include <hip/hip_bf16.h>

// ---------------------------------------------------------------------------
// MemoryEfficientAttention for MI455X (gfx1250, wave32, WMMA bf16 path)
// B=2, N=2048, C=1024, H=16, D=64, CHUNK=128 (softmax per key-chunk)
// Async LDS staging of K/V chunks via global_load_async_to_lds_b128.
// ---------------------------------------------------------------------------

typedef __bf16 bf16;
typedef __attribute__((ext_vector_type(16))) __bf16 v16bf;
typedef __attribute__((ext_vector_type(8)))  float  v8f;

#define BATCH 2
#define NTOK  2048
#define CDIM  1024
#define HEADS 16
#define DHEAD 64
#define CHUNK 128
#define MROWS (BATCH * NTOK)   // 4096
#define KPAD  72               // 64 d + 8 pad  (k-chunk LDS row, 144 B)
#define VPAD  136              // 128 keys + 8 pad (v-chunk / P LDS row, 272 B)

// ---- helpers --------------------------------------------------------------

__device__ __forceinline__ unsigned short bf16_bits(float f) {
  union { float f; unsigned u; } x; x.f = f;
  return (unsigned short)((x.u + 0x7FFFu + ((x.u >> 16) & 1u)) >> 16); // RNE
}
__device__ __forceinline__ bf16 to_bf16(float f) {
  union { unsigned short s; bf16 b; } y; y.s = bf16_bits(f);
  return y.b;
}

// Per-lane WMMA fragment load (A operand, or B operand from a transposed
// row-major source; identical per-lane layout on CDNA5 wave32):
//   lane l holds entity row (l & 15); elements at K offsets
//   kbase + 8*(l>>4) + {0..7}  and  kbase + 16 + 8*(l>>4) + {0..7}
__device__ __forceinline__ v16bf load_frag(const bf16* rowPtr, int kbase, int half) {
  union { v16bf v; uint4 u[2]; } t;
  t.u[0] = *(const uint4*)(rowPtr + kbase + 8 * half);
  t.u[1] = *(const uint4*)(rowPtr + kbase + 16 + 8 * half);
  return t.v;
}

__device__ __forceinline__ v8f wmma_bf16(v16bf a, v16bf b, v8f c) {
  return __builtin_amdgcn_wmma_f32_16x16x32_bf16(
      /*neg_a=*/false, a, /*neg_b=*/false, b,
      /*c_mod=*/(short)0, c, /*reuse_a=*/false, /*reuse_b=*/false);
}

// Async copy 16 bytes global -> LDS (ASYNCcnt-tracked, no VGPR staging)
__device__ __forceinline__ void async_g2l_b128(void* lds, const void* gaddr) {
  unsigned ldsOff = (unsigned)(unsigned long long)lds;   // flat low 32 = LDS offset
  asm volatile("global_load_async_to_lds_b128 %0, %1, off"
               :: "v"(ldsOff), "v"(gaddr) : "memory");
}

// ---- conversion kernels ---------------------------------------------------

__global__ void cvt_copy_bf16(const float* __restrict__ src,
                              bf16* __restrict__ dst, int n) {
  int i = blockIdx.x * blockDim.x + threadIdx.x;
  if (i < n) dst[i] = to_bf16(src[i]);
}

// src[r][c] (rows x cols, f32) -> dst[c][r] (bf16), scaled
__global__ void cvt_transpose_bf16(const float* __restrict__ src,
                                   bf16* __restrict__ dst,
                                   int rows, int cols, float scale) {
  int i = blockIdx.x * blockDim.x + threadIdx.x;
  if (i < rows * cols) {
    int r = i / cols, c = i % cols;
    dst[(size_t)c * rows + r] = to_bf16(src[i] * scale);
  }
}

// ---- QKV projection GEMM --------------------------------------------------
// C[m,n] = X[m,:] . Wt[n,:]   (Wt transposed weight, bf16)
// grid: (MROWS/128, CDIM/128, 3) ; block 256 (8 waves); wave tile 32x64
__global__ __launch_bounds__(256) void qkv_gemm(
    const bf16* __restrict__ X,
    const bf16* __restrict__ Wqt, const bf16* __restrict__ Wkt,
    const bf16* __restrict__ Wvt,
    bf16* __restrict__ Q,  /* [B,H,N,D] */
    bf16* __restrict__ K,  /* [B,H,N,D] */
    bf16* __restrict__ VT  /* [B,H,D,N] */) {
  const int mode = blockIdx.z;
  const bf16* Wt = (mode == 0) ? Wqt : ((mode == 1) ? Wkt : Wvt);

  const int lane = threadIdx.x & 31;
  const int wave = threadIdx.x >> 5;       // 0..7
  const int wm = wave >> 1;                // 0..3
  const int wn = wave & 1;                 // 0..1
  const int l16 = lane & 15, half = lane >> 4;
  const int mBase = blockIdx.x * 128 + wm * 32;
  const int nBase = blockIdx.y * 128 + wn * 64;

  v8f acc[2][4] = {};

  for (int k0 = 0; k0 < CDIM; k0 += 32) {
    v16bf a[2], b[4];
#pragma unroll
    for (int i = 0; i < 2; ++i)
      a[i] = load_frag(X + (size_t)(mBase + i * 16 + l16) * CDIM, k0, half);
#pragma unroll
    for (int j = 0; j < 4; ++j)
      b[j] = load_frag(Wt + (size_t)(nBase + j * 16 + l16) * CDIM, k0, half);
#pragma unroll
    for (int i = 0; i < 2; ++i)
#pragma unroll
      for (int j = 0; j < 4; ++j)
        acc[i][j] = wmma_bf16(a[i], b[j], acc[i][j]);
  }

  // D layout: lane holds col n = tile + l16, rows m = r + 8*half.
  // nBase is a multiple of 64 => head h is uniform per wave.
  const int h = nBase >> 6;
  if (mode != 2) {
    bf16* __restrict__ dst = (mode == 0) ? Q : K;
#pragma unroll
    for (int i = 0; i < 2; ++i)
#pragma unroll
      for (int j = 0; j < 4; ++j) {
        const int d = (nBase & 63) + j * 16 + l16;  // 0..63
#pragma unroll
        for (int r = 0; r < 8; ++r) {
          const int m = mBase + i * 16 + r + 8 * half;
          const int bIdx = m >> 11, tok = m & (NTOK - 1);
          dst[(((size_t)bIdx * HEADS + h) * NTOK + tok) * DHEAD + d] =
              to_bf16(acc[i][j][r]);
        }
      }
  } else {
#pragma unroll
    for (int i = 0; i < 2; ++i)
#pragma unroll
      for (int j = 0; j < 4; ++j) {
        const int d = (nBase & 63) + j * 16 + l16;
#pragma unroll
        for (int r = 0; r < 8; ++r) {
          const int m = mBase + i * 16 + r + 8 * half;
          const int bIdx = m >> 11, tok = m & (NTOK - 1);
          VT[(((size_t)bIdx * HEADS + h) * DHEAD + d) * NTOK + tok] =
              to_bf16(acc[i][j][r]);
        }
      }
  }
}

// ---- attention kernel -----------------------------------------------------
// grid: (N/64, B*H) ; block 128 = 4 waves; each wave owns 16 query rows.
// Per key-chunk: async-stage K/V chunk into LDS (once per WG), S^T via WMMA,
// per-chunk softmax, P through wave-private LDS, O += P@V via WMMA.
__global__ __launch_bounds__(128) void attn_chunked(
    const bf16* __restrict__ Q,  /* [B,H,N,D], pre-scaled via Wq */
    const bf16* __restrict__ K,  /* [B,H,N,D] */
    const bf16* __restrict__ VT, /* [B,H,D,N] */
    bf16* __restrict__ O         /* [B,N,C]   */) {
  __shared__ __align__(16) bf16 kbuf[CHUNK * KPAD];   // 18.0 KB
  __shared__ __align__(16) bf16 vbuf[DHEAD * VPAD];   // 17.0 KB
  __shared__ __align__(16) bf16 plds[4][16 * VPAD];   // 17.0 KB

  const int tid  = threadIdx.x;         // 0..127
  const int lane = tid & 31;
  const int wave = tid >> 5;            // 0..3
  const int l16 = lane & 15, half = lane >> 4;
  const int bh = blockIdx.y;            // b*HEADS + h
  const int b = bh >> 4, h = bh & 15;
  const int qBase = blockIdx.x * 64 + wave * 16;

  const bf16* qhead = Q + (size_t)bh * NTOK * DHEAD;
  const bf16* khead = K + (size_t)bh * NTOK * DHEAD;
  const bf16* vth   = VT + (size_t)bh * DHEAD * NTOK;

  // q fragments: serve as WMMA *B* operand for S^T = k . q^T
  const bf16* qrow = qhead + (size_t)(qBase + l16) * DHEAD;
  v16bf qf0 = load_frag(qrow, 0, half);
  v16bf qf1 = load_frag(qrow, 32, half);

  v8f oacc[4] = {};
  bf16* P = plds[wave];  // wave-private 16 x VPAD tile, P[query][key]

  const int vrowIdx = tid >> 1;         // 0..63 (d row of V^T chunk)
  const int vhalf   = tid & 1;          // which 128B half of the 256B row

  for (int c = 0; c < NTOK / CHUNK; ++c) {
    const int kBase = c * CHUNK;

    // ---- async-stage this chunk's K (128x64) and V^T (64x128) into LDS ----
    {
      const bf16* kg = khead + (size_t)(kBase + tid) * DHEAD;       // my row
      bf16*       kl = kbuf + tid * KPAD;
      const bf16* vg = vth + (size_t)vrowIdx * NTOK + kBase + vhalf * 64;
      bf16*       vl = vbuf + vrowIdx * VPAD + vhalf * 64;
#pragma unroll
      for (int i = 0; i < 8; ++i)
        async_g2l_b128(kl + i * 8, kg + i * 8);                     // 16 B each
#pragma unroll
      for (int i = 0; i < 8; ++i)
        async_g2l_b128(vl + i * 8, vg + i * 8);
      asm volatile("s_wait_asynccnt 0" ::: "memory");
    }
    __syncthreads();

    // ---- S^T tiles: rows = key (r + 8*half), cols = query (l16) ----
    v8f st[8];
#pragma unroll
    for (int j = 0; j < 8; ++j) {
      const bf16* krow = kbuf + (j * 16 + l16) * KPAD;
      v16bf kf0 = load_frag(krow, 0, half);
      v16bf kf1 = load_frag(krow, 32, half);
      v8f s = {};
      s = wmma_bf16(kf0, qf0, s);
      s = wmma_bf16(kf1, qf1, s);
      st[j] = s;
    }

    // ---- per-chunk softmax over 128 keys for query column l16 ----
    float mx = -3.0e38f;
#pragma unroll
    for (int j = 0; j < 8; ++j)
#pragma unroll
      for (int r = 0; r < 8; ++r) mx = fmaxf(mx, st[j][r]);
    mx = fmaxf(mx, __shfl_xor(mx, 16, 32));

    float sum = 0.0f;
#pragma unroll
    for (int j = 0; j < 8; ++j)
#pragma unroll
      for (int r = 0; r < 8; ++r) {
        float e = __expf(st[j][r] - mx);
        st[j][r] = e;
        sum += e;
      }
    sum += __shfl_xor(sum, 16, 32);
    const float inv = __frcp_rn(sum);

    // ---- write normalized P to LDS as P[query=l16][key] (b32 pair packs) --
#pragma unroll
    for (int j = 0; j < 8; ++j) {
      const int kb = j * 16 + 8 * half;
#pragma unroll
      for (int r = 0; r < 8; r += 2) {
        unsigned lo = bf16_bits(st[j][r] * inv);
        unsigned hi = bf16_bits(st[j][r + 1] * inv);
        *(unsigned*)&P[l16 * VPAD + kb + r] = (hi << 16) | lo;
      }
    }
    asm volatile("s_wait_dscnt 0" ::: "memory");  // wave-private P visible

    // ---- O += P @ V : A = P (16q x 32k), B = V^T rows (d, keys) ----
#pragma unroll
    for (int kg = 0; kg < 4; ++kg) {
      v16bf pa = load_frag(P + l16 * VPAD, kg * 32, half);
#pragma unroll
      for (int j = 0; j < 4; ++j) {
        const bf16* vrow = vbuf + (j * 16 + l16) * VPAD;
        v16bf vf = load_frag(vrow, kg * 32, half);
        oacc[j] = wmma_bf16(pa, vf, oacc[j]);
      }
    }
    __syncthreads();  // all waves done reading kbuf/vbuf before next stage
  }

  // store O: lane holds (row query = qBase + r + 8*half, col d = j*16+l16)
#pragma unroll
  for (int j = 0; j < 4; ++j) {
    const int d = j * 16 + l16;
#pragma unroll
    for (int r = 0; r < 8; ++r) {
      const int qrowi = qBase + r + 8 * half;
      O[((size_t)b * NTOK + qrowi) * CDIM + h * DHEAD + d] = to_bf16(oacc[j][r]);
    }
  }
}

// ---- output projection GEMM (adds bias, f32 out) --------------------------
__global__ __launch_bounds__(256) void out_gemm(
    const bf16* __restrict__ A,   /* [B*N, C] bf16 */
    const bf16* __restrict__ Wpt, /* [C, C] transposed bf16 */
    const float* __restrict__ bias,
    float* __restrict__ Out /* [B*N, C] f32 */) {
  const int lane = threadIdx.x & 31;
  const int wave = threadIdx.x >> 5;
  const int wm = wave >> 1, wn = wave & 1;
  const int l16 = lane & 15, half = lane >> 4;
  const int mBase = blockIdx.x * 128 + wm * 32;
  const int nBase = blockIdx.y * 128 + wn * 64;

  v8f acc[2][4] = {};
  for (int k0 = 0; k0 < CDIM; k0 += 32) {
    v16bf a[2], bf[4];
#pragma unroll
    for (int i = 0; i < 2; ++i)
      a[i] = load_frag(A + (size_t)(mBase + i * 16 + l16) * CDIM, k0, half);
#pragma unroll
    for (int j = 0; j < 4; ++j)
      bf[j] = load_frag(Wpt + (size_t)(nBase + j * 16 + l16) * CDIM, k0, half);
#pragma unroll
    for (int i = 0; i < 2; ++i)
#pragma unroll
      for (int j = 0; j < 4; ++j)
        acc[i][j] = wmma_bf16(a[i], bf[j], acc[i][j]);
  }

#pragma unroll
  for (int i = 0; i < 2; ++i)
#pragma unroll
    for (int j = 0; j < 4; ++j) {
      const int col = nBase + j * 16 + l16;
      const float bv = bias[col];
#pragma unroll
      for (int r = 0; r < 8; ++r) {
        const int m = mBase + i * 16 + r + 8 * half;
        Out[(size_t)m * CDIM + col] = acc[i][j][r] + bv;
      }
    }
}

// ---------------------------------------------------------------------------

extern "C" void kernel_launch(void* const* d_in, const int* in_sizes, int n_in,
                              void* d_out, int out_size, void* d_ws, size_t ws_size,
                              hipStream_t stream) {
  const float* x  = (const float*)d_in[0];
  const float* Wq = (const float*)d_in[1];
  const float* Wk = (const float*)d_in[2];
  const float* Wv = (const float*)d_in[3];
  const float* Wp = (const float*)d_in[4];
  const float* bp = (const float*)d_in[5];
  float* out = (float*)d_out;

  char* ws = (char*)d_ws;
  const size_t MB = 1u << 20;
  bf16* xbf = (bf16*)(ws + 0 * MB);    // 8 MB  [B*N, C]
  bf16* wqt = (bf16*)(ws + 8 * MB);    // 2 MB  [C, C]^T (scaled)
  bf16* wkt = (bf16*)(ws + 10 * MB);   // 2 MB
  bf16* wvt = (bf16*)(ws + 12 * MB);   // 2 MB
  bf16* wpt = (bf16*)(ws + 14 * MB);   // 2 MB
  bf16* q   = (bf16*)(ws + 16 * MB);   // 8 MB  [B,H,N,D]
  bf16* k   = (bf16*)(ws + 24 * MB);   // 8 MB  [B,H,N,D]
  bf16* vt  = (bf16*)(ws + 32 * MB);   // 8 MB  [B,H,D,N]
  bf16* ao  = (bf16*)(ws + 40 * MB);   // 8 MB  [B,N,C]

  const float scale = 0.125f;  // D^-0.5 = 64^-0.5, folded into Wq

  cvt_copy_bf16<<<(MROWS * CDIM) / 256, 256, 0, stream>>>(x, xbf, MROWS * CDIM);
  cvt_transpose_bf16<<<(CDIM * CDIM) / 256, 256, 0, stream>>>(Wq, wqt, CDIM, CDIM, scale);
  cvt_transpose_bf16<<<(CDIM * CDIM) / 256, 256, 0, stream>>>(Wk, wkt, CDIM, CDIM, 1.0f);
  cvt_transpose_bf16<<<(CDIM * CDIM) / 256, 256, 0, stream>>>(Wv, wvt, CDIM, CDIM, 1.0f);
  cvt_transpose_bf16<<<(CDIM * CDIM) / 256, 256, 0, stream>>>(Wp, wpt, CDIM, CDIM, 1.0f);

  qkv_gemm<<<dim3(MROWS / 128, CDIM / 128, 3), 256, 0, stream>>>(
      xbf, wqt, wkt, wvt, q, k, vt);

  attn_chunked<<<dim3(NTOK / 64, BATCH * HEADS), 128, 0, stream>>>(q, k, vt, ao);

  out_gemm<<<dim3(MROWS / 128, CDIM / 128), 256, 0, stream>>>(ao, wpt, bp, out);
}
